// NonLocalBlock_35811437314407
// MI455X (gfx1250) — matrix operand order repaired and checked by reading, not compiled, not run on hardware
//
#include <hip/hip_runtime.h>
#include <hip/hip_bf16.h>

// Problem constants (from reference)
#define B_   2
#define C_   256
#define CH_  128
#define H_   64
#define W_   64
#define HW_  (H_ * W_)     // 4096
#define L_   (CH_ * H_)    // 8192
#define D_   64            // attention head dim (= W)

typedef __bf16 bf16_t;
typedef __bf16 v16bf __attribute__((ext_vector_type(16)));
typedef float  v8f   __attribute__((ext_vector_type(8)));
typedef unsigned int v4u __attribute__((ext_vector_type(4)));
typedef int v4i __attribute__((ext_vector_type(4)));
typedef int v8i __attribute__((ext_vector_type(8)));

union frag16 { v16bf v; uint4 u[2]; };

// Native RNE f32 -> bf16 (v_cvt_pk_bf16_f32 on gfx1250)
static __device__ __forceinline__ unsigned short f32_bf16(float f) {
  return __builtin_bit_cast(unsigned short, static_cast<bf16_t>(f));
}

// A-fragment (16x32, 16-bit) from an f32 row: 16 contiguous elems per lane:
// k = half*16 + e
static __device__ __forceinline__ v16bf load_a_f32(const float* __restrict__ row,
                                                   int kc, int half) {
  const float4* p0 = (const float4*)(row + kc * 32 + half * 16);
  const float4* p1 = (const float4*)(row + kc * 32 + half * 16 + 8);
  float4 c0 = p0[0], c1 = p0[1], c2 = p1[0], c3 = p1[1];
  v16bf a;
  a[0]  = (bf16_t)c0.x; a[1]  = (bf16_t)c0.y; a[2]  = (bf16_t)c0.z; a[3]  = (bf16_t)c0.w;
  a[4]  = (bf16_t)c1.x; a[5]  = (bf16_t)c1.y; a[6]  = (bf16_t)c1.z; a[7]  = (bf16_t)c1.w;
  a[8]  = (bf16_t)c2.x; a[9]  = (bf16_t)c2.y; a[10] = (bf16_t)c2.z; a[11] = (bf16_t)c2.w;
  a[12] = (bf16_t)c3.x; a[13] = (bf16_t)c3.y; a[14] = (bf16_t)c3.z; a[15] = (bf16_t)c3.w;
  return a;
}

// A-fragment from a bf16 row (two 16B vector loads)
static __device__ __forceinline__ v16bf load_a_bf16(const unsigned short* __restrict__ row,
                                                    int kc, int half) {
  frag16 f;
  f.u[0] = *(const uint4*)(row + kc * 32 + half * 16);
  f.u[1] = *(const uint4*)(row + kc * 32 + half * 16 + 8);
  return f.v;
}

// B-fragment (32x16, 16-bit) when the fragment's 16 K-values for this lane
// are contiguous in memory: lanes 0-15 hold K=0-15, lanes 16-31 K=16-31.
static __device__ __forceinline__ v16bf load_b_contig(const unsigned short* __restrict__ p) {
  frag16 f;
  f.u[0] = *(const uint4*)p;
  f.u[1] = *(const uint4*)(p + 8);
  return f.v;
}

// Tensor Data Mover: 2D bf16 tile (tile_d1 rows x tile_d0 elems, row stride
// stride0) from global to LDS.  D# layout per CDNA5 ISA.
// 6-arg builtin form (amdgpu-toolchain / clang-23).
static __device__ __forceinline__ void tdm_load_2d_bf16(
    unsigned lds_off, const void* gaddr, unsigned tensor_d0, unsigned tensor_d1,
    unsigned tile_d0, unsigned tile_d1, unsigned stride0) {
  unsigned long long ga = (unsigned long long)(size_t)gaddr;
  v4u g0;
  g0[0] = 1u;                                          // count=1, user D#
  g0[1] = lds_off;                                     // lds_addr (bytes)
  g0[2] = (unsigned)(ga & 0xFFFFFFFFu);                // global_addr[31:0]
  g0[3] = (unsigned)((ga >> 32) & 0x01FFFFFFu)         // global_addr[56:32]
        | (2u << 30);                                  // type=2 ("image")
  v8i g1;
  g1[0] = (int)(1u << 16);                             // data_size=1 (2B), mask=0
  g1[1] = (int)((tensor_d0 & 0xFFFFu) << 16);          // tensor_dim0[15:0]
  g1[2] = (int)(((tensor_d0 >> 16) & 0xFFFFu)
        |        ((tensor_d1 & 0xFFFFu) << 16));       // dim0 hi | dim1 lo
  g1[3] = (int)(((tensor_d1 >> 16) & 0xFFFFu)
        |        ((tile_d0 & 0xFFFFu) << 16));         // dim1 hi | tile_dim0
  g1[4] = (int)(tile_d1 & 0xFFFFu);                    // tile_dim1, tile_dim2=0
  g1[5] = (int)stride0;                                // tensor_dim0_stride lo
  g1[6] = 0;                                           // stride hi | dim1_stride lo
  g1[7] = 0;
  v4i z4 = {0, 0, 0, 0};                               // groups 2/3 unused (2D)
  v8i z8 = {0, 0, 0, 0, 0, 0, 0, 0};
  __builtin_amdgcn_tensor_load_to_lds(g0, g1, z4, z4, z8, 0);
}

// ---------------------------------------------------------------------------
// Kernel 1: Q/K/V projections.  One block = one (batch, 16-col HW strip);
// stages X strip in LDS (bf16, transposed) shared by all 24 output tiles.
// ---------------------------------------------------------------------------
__global__ __launch_bounds__(128) void qkv_proj(
    const float* __restrict__ inp,
    const float* __restrict__ w0, const float* __restrict__ bb0,
    const float* __restrict__ w1, const float* __restrict__ bb1,
    const float* __restrict__ w2, const float* __restrict__ bb2,
    unsigned short* __restrict__ qws, unsigned short* __restrict__ kws,
    unsigned short* __restrict__ vws) {
  const int wid  = threadIdx.x >> 5;
  const int lane = threadIdx.x & 31;
  const int half = lane >> 4;
  const int lan  = lane & 15;

  const int b  = blockIdx.x >> 8;   // batch
  const int nt = blockIdx.x & 255;  // HW 16-col strip

  const float* X = inp + (size_t)b * C_ * HW_;

  __shared__ __align__(16) unsigned short Xt[16 * C_];  // (n, k) transposed, 8 KB

  #pragma unroll
  for (int i = 0; i < 32; ++i) {
    int idx = threadIdx.x + i * 128;     // 0..4095
    int k = idx >> 4, n = idx & 15;
    Xt[n * C_ + k] = f32_bf16(X[(size_t)k * HW_ + nt * 16 + n]);
  }
  __syncthreads();

  #pragma unroll
  for (int j = 0; j < 6; ++j) {
    int tile  = wid * 6 + j;     // 0..23
    int which = tile >> 3;       // theta / phi / g
    int mt    = tile & 7;        // CH 16-row tile
    const float* Wm = (which == 0) ? w0 : (which == 1) ? w1 : w2;
    const float* Bs = (which == 0) ? bb0 : (which == 1) ? bb1 : bb2;
    unsigned short* Out =
        ((which == 0) ? qws : (which == 1) ? kws : vws) + (size_t)b * CH_ * HW_;

    v8f acc = {};
    #pragma unroll
    for (int kc = 0; kc < C_ / 32; ++kc) {
      v16bf a  = load_a_f32(Wm + (size_t)(mt * 16 + lan) * C_, kc, half);
      v16bf bm = load_b_contig(&Xt[lan * C_ + kc * 32 + half * 16]);
      acc = __builtin_amdgcn_wmma_f32_16x16x32_bf16(false, a, false, bm,
                                                    (short)0, acc, false, false);
    }
    #pragma unroll
    for (int v = 0; v < 8; ++v) {
      int m = mt * 16 + v + 8 * half;
      int n = nt * 16 + lan;
      Out[(size_t)m * HW_ + n] = f32_bf16(acc[v] + Bs[m]);
    }
  }
}

// ---------------------------------------------------------------------------
// Kernel 2: flash attention, L=8192, D=64, no scaling (per reference).
// 4 waves/block; wave owns 16 query rows; loop over key tiles of 32.
// K tile arrives via the Tensor Data Mover; V tile is transposed on store.
// ---------------------------------------------------------------------------
__global__ __launch_bounds__(128) void flash_attn(
    const unsigned short* __restrict__ qws,
    const unsigned short* __restrict__ kws,
    const unsigned short* __restrict__ vws,
    float* __restrict__ ows) {
  const int wid  = threadIdx.x >> 5;
  const int lane = threadIdx.x & 31;
  const int half = lane >> 4;
  const int lan  = lane & 15;

  const int b   = blockIdx.x >> 7;       // 2 batches
  const int qb  = blockIdx.x & 127;      // 128 query blocks of 64 rows
  const int qr0 = qb * 64 + wid * 16;

  const unsigned short* Q = qws + (size_t)b * L_ * D_;
  const unsigned short* K = kws + (size_t)b * L_ * D_;
  const unsigned short* V = vws + (size_t)b * L_ * D_;

  __shared__ __align__(16) unsigned short Kt[32 * D_];   // (key, d)  4 KB
  __shared__ __align__(16) unsigned short Vt[D_ * 32];   // (d, key)  4 KB (transposed)
  __shared__ __align__(16) unsigned short Pb[4][16 * 32];
  __shared__ __align__(16) float Ss[4][16 * 32];
  __shared__ float mrow[4][16], lrow[4][16], srow[4][16];

  // Q A-fragments for this wave's 16 rows, both 32-wide k-chunks of D=64
  const unsigned short* Qrow = Q + (size_t)(qr0 + lan) * D_;
  v16bf aq0 = load_a_bf16(Qrow, 0, half);
  v16bf aq1 = load_a_bf16(Qrow, 1, half);

  v8f O0 = {}, O1 = {}, O2 = {}, O3 = {};
  if (lane < 16) { mrow[wid][lane] = -1e30f; lrow[wid][lane] = 0.0f; }

  const unsigned kt_lds = (unsigned)(size_t)&Kt[0];      // LDS byte address

  for (int t = 0; t < L_ / 32; ++t) {
    // K tile: TDM DMA (32 rows x 64 bf16, row stride D) global -> LDS.
    if (wid == 0) {
      tdm_load_2d_bf16(kt_lds, K + (size_t)t * 32 * D_,
                       /*tensor_d0=*/D_, /*tensor_d1=*/L_,
                       /*tile_d0=*/D_, /*tile_d1=*/32, /*stride0=*/D_);
    }
    // V tile: transpose on store (TDM cannot transpose).
    const uint4* Vg = (const uint4*)(V + (size_t)t * 32 * D_);
    #pragma unroll
    for (int i = 0; i < 2; ++i) {
      int idx = threadIdx.x + i * 128;        // 0..255 uint4s
      uint4 gv = Vg[idx];
      const unsigned short* pv = (const unsigned short*)&gv;
      int jk = idx >> 3, d0 = (idx & 7) * 8;  // key, d-chunk
      #pragma unroll
      for (int q2 = 0; q2 < 8; ++q2) Vt[(d0 + q2) * 32 + jk] = pv[q2];
    }
    if (t + 1 < L_ / 32) {                     // global_prefetch_b8 next V tile
      __builtin_prefetch(V + (size_t)(t + 1) * 32 * D_, 0, 0);
    }
    if (wid == 0) __builtin_amdgcn_s_wait_tensorcnt(0);
    __syncthreads();

    // S = Q * K^T for this wave's 16 rows x 32 keys (4 WMMAs)
    #pragma unroll
    for (int nt2 = 0; nt2 < 2; ++nt2) {
      v8f s = {};
      #pragma unroll
      for (int kc = 0; kc < 2; ++kc) {
        v16bf bk = load_b_contig(&Kt[(nt2 * 16 + lan) * D_ + kc * 32 + half * 16]);
        s = __builtin_amdgcn_wmma_f32_16x16x32_bf16(
                false, (kc == 0) ? aq0 : aq1, false, bk, (short)0, s,
                false, false);
      }
      #pragma unroll
      for (int v = 0; v < 8; ++v)
        Ss[wid][(v + 8 * half) * 32 + nt2 * 16 + lan] = s[v];
    }

    // Online softmax (fp32): lane -> row = lan, column half = half
    float sv[16];
    const float4* Sp = (const float4*)&Ss[wid][lan * 32 + half * 16];
    #pragma unroll
    for (int i = 0; i < 4; ++i) {
      float4 x = Sp[i];
      sv[4 * i] = x.x; sv[4 * i + 1] = x.y; sv[4 * i + 2] = x.z; sv[4 * i + 3] = x.w;
    }
    float mloc = sv[0];
    #pragma unroll
    for (int i = 1; i < 16; ++i) mloc = fmaxf(mloc, sv[i]);
    mloc = fmaxf(mloc, __shfl_xor(mloc, 16, 32));
    float mold = mrow[wid][lan];
    float mnew = fmaxf(mold, mloc);
    float sc   = __expf(mold - mnew);
    float suml = 0.0f;
    union { uint4 u[2]; unsigned short h[16]; } pk;
    #pragma unroll
    for (int i = 0; i < 16; ++i) {
      float p = __expf(sv[i] - mnew);
      pk.h[i] = f32_bf16(p);
      suml += p;
    }
    suml += __shfl_xor(suml, 16, 32);
    uint4* Pp = (uint4*)&Pb[wid][lan * 32 + half * 16];
    Pp[0] = pk.u[0]; Pp[1] = pk.u[1];
    if (lane < 16) {
      mrow[wid][lan] = mnew;
      lrow[wid][lan] = lrow[wid][lan] * sc + suml;
      srow[wid][lan] = sc;
    }

    // Rescale O accumulators by per-row exp(m_old - m_new)
    #pragma unroll
    for (int v = 0; v < 8; ++v) {
      float f = srow[wid][v + 8 * half];
      O0[v] *= f; O1[v] *= f; O2[v] *= f; O3[v] *= f;
    }

    // P (A-fragment) and P*V accumulation (4 WMMAs over D)
    v16bf ap = load_a_bf16(&Pb[wid][lan * 32], 0, half);
    #pragma unroll
    for (int c = 0; c < 4; ++c) {
      v16bf bv = load_b_contig(&Vt[(c * 16 + lan) * 32 + half * 16]);
      v8f& Oc = (c == 0) ? O0 : (c == 1) ? O1 : (c == 2) ? O2 : O3;
      Oc = __builtin_amdgcn_wmma_f32_16x16x32_bf16(false, ap, false, bv,
                                                   (short)0, Oc, false, false);
    }
    __syncthreads();   // before next tile overwrites Kt/Vt
  }

  // Final normalization by row sums, store fp32 O (layout (L, D))
  float* Ob = ows + (size_t)b * L_ * D_;
  #pragma unroll
  for (int v = 0; v < 8; ++v) {
    float linv = 1.0f / lrow[wid][v + 8 * half];
    size_t r = (size_t)(qr0 + v + 8 * half) * D_;
    Ob[r +  0 + lan] = O0[v] * linv;
    Ob[r + 16 + lan] = O1[v] * linv;
    Ob[r + 32 + lan] = O2[v] * linv;
    Ob[r + 48 + lan] = O3[v] * linv;
  }
}

// ---------------------------------------------------------------------------
// Kernel 3: out = w_out(256x128) @ o(128x4096) + b_out + inp  (fp32 out)
// One block = one (batch, 16-col strip); O strip staged transposed in LDS.
// ---------------------------------------------------------------------------
__global__ __launch_bounds__(128) void out_proj(
    const float* __restrict__ ows, const float* __restrict__ w_out,
    const float* __restrict__ b_out, const float* __restrict__ inp,
    float* __restrict__ out) {
  const int wid  = threadIdx.x >> 5;
  const int lane = threadIdx.x & 31;
  const int half = lane >> 4;
  const int lan  = lane & 15;

  const int b  = blockIdx.x >> 8;
  const int nt = blockIdx.x & 255;

  const float* Ob = ows + (size_t)b * L_ * D_;   // == (CH_, HW_) row-major

  __shared__ __align__(16) unsigned short Ot[16 * CH_];  // (n, k) transposed, 4 KB

  #pragma unroll
  for (int i = 0; i < 16; ++i) {
    int idx = threadIdx.x + i * 128;     // 0..2047
    int k = idx >> 4, n = idx & 15;
    Ot[n * CH_ + k] = f32_bf16(Ob[(size_t)k * HW_ + nt * 16 + n]);
  }
  __syncthreads();

  #pragma unroll
  for (int j = 0; j < 4; ++j) {
    int mt = wid * 4 + j;                // 0..15
    v8f acc = {};
    #pragma unroll
    for (int kc = 0; kc < CH_ / 32; ++kc) {
      v16bf a  = load_a_f32(w_out + (size_t)(mt * 16 + lan) * CH_, kc, half);
      v16bf bm = load_b_contig(&Ot[lan * CH_ + kc * 32 + half * 16]);
      acc = __builtin_amdgcn_wmma_f32_16x16x32_bf16(false, a, false, bm,
                                                    (short)0, acc, false, false);
    }
    #pragma unroll
    for (int v = 0; v < 8; ++v) {
      int m = mt * 16 + v + 8 * half;
      int n = nt * 16 + lan;
      size_t idx = (size_t)b * C_ * HW_ + (size_t)m * HW_ + n;
      out[idx] = acc[v] + b_out[m] + inp[idx];
    }
  }
}

// ---------------------------------------------------------------------------
extern "C" void kernel_launch(void* const* d_in, const int* in_sizes, int n_in,
                              void* d_out, int out_size, void* d_ws,
                              size_t ws_size, hipStream_t stream) {
  const float* inp     = (const float*)d_in[0];
  const float* w_theta = (const float*)d_in[1];
  const float* b_theta = (const float*)d_in[2];
  const float* w_phi   = (const float*)d_in[3];
  const float* b_phi   = (const float*)d_in[4];
  const float* w_g     = (const float*)d_in[5];
  const float* b_g     = (const float*)d_in[6];
  const float* w_out   = (const float*)d_in[7];
  const float* b_out   = (const float*)d_in[8];

  // Workspace: Q/K/V in bf16 (2 MB each), O in fp32 (4 MB) -> 10 MB total
  unsigned short* qws = (unsigned short*)d_ws;
  unsigned short* kws = qws + (size_t)B_ * L_ * D_;
  unsigned short* vws = kws + (size_t)B_ * L_ * D_;
  float*          owsp = (float*)(vws + (size_t)B_ * L_ * D_);

  // 2 batches * 256 HW strips
  qkv_proj<<<512, 128, 0, stream>>>(inp, w_theta, b_theta, w_phi, b_phi,
                                    w_g, b_g, qws, kws, vws);
  // 2 batches * 128 query blocks
  flash_attn<<<256, 128, 0, stream>>>(qws, kws, vws, owsp);
  // 2 batches * 256 HW strips
  out_proj<<<512, 128, 0, stream>>>(owsp, w_out, b_out, inp, (float*)d_out);
}
